// align_90151363543690
// MI455X (gfx1250) — matrix operand order repaired and checked by reading, not compiled, not run on hardware
//
#include <hip/hip_runtime.h>
#include <hip/hip_bf16.h>
#include <stdint.h>

typedef __attribute__((ext_vector_type(16))) __bf16 v16bf;
typedef __attribute__((ext_vector_type(8)))  float  v8f;

union FragU { uint4 q[2]; v16bf v; };

__device__ __forceinline__ unsigned short f2bf(float f) {
  unsigned u = __builtin_bit_cast(unsigned, f);
  u += 0x7FFFu + ((u >> 16) & 1u);          // round-to-nearest-even
  return (unsigned short)(u >> 16);
}
__device__ __forceinline__ float bf2f(unsigned short h) {
  unsigned u = ((unsigned)h) << 16;
  return __builtin_bit_cast(float, u);
}
__device__ __forceinline__ int iclamp(int v, int lo, int hi) {
  return v < lo ? lo : (v > hi ? hi : v);
}

// Direct global -> LDS async copy (CDNA5 TDM-adjacent path, ASYNCcnt-tracked).
// Same INST_OFFSET applies to both the LDS and the global address (ISA 10.x),
// so one base pair + offset:16 moves a full 32B half-row.
__device__ __forceinline__ void async_b128(unsigned lds_off, const void* g) {
  asm volatile("global_load_async_to_lds_b128 %0, %1, off"
               :: "v"(lds_off), "v"(g) : "memory");
}
__device__ __forceinline__ void async_b128_o16(unsigned lds_off, const void* g) {
  asm volatile("global_load_async_to_lds_b128 %0, %1, off offset:16"
               :: "v"(lds_off), "v"(g) : "memory");
}
__device__ __forceinline__ void wait_async_le2() {
  asm volatile("s_wait_asynccnt 0x2" ::: "memory");
}
__device__ __forceinline__ void wait_async_0() {
  asm volatile("s_wait_asynccnt 0x0" ::: "memory");
}

// ---------------------------------------------------------------------------
// Pack concat(ir, vis) NCHW f32 -> NHWC bf16 [16384][512]
// ---------------------------------------------------------------------------
__global__ __launch_bounds__(256)
void pack_fused_kernel(const float* __restrict__ ir, const float* __restrict__ vis,
                       unsigned short* __restrict__ out) {
  int i = blockIdx.x * 256 + threadIdx.x;        // over 16384*512
  if (i >= 16384 * 512) return;
  int c2 = i & 511;
  int p  = i >> 9;                               // b*4096 + h*64 + w
  int b  = p >> 12;
  int hw = p & 4095;
  const float* src = (c2 < 256) ? ir : vis;
  int c = c2 & 255;
  float v = src[(((long long)b * 256 + c) << 12) + hw];
  out[i] = f2bf(v);
}

// ---------------------------------------------------------------------------
// Weight transpose: w [Cout][Cin][3][3] f32 -> wt [CoutPad][9*Cin] bf16,
// wt[co][kidx*Cin + ci] = w[co][ci][kidx]; rows co>=Cout zero-filled.
// ---------------------------------------------------------------------------
__global__ __launch_bounds__(256)
void pack_wt_kernel(const float* __restrict__ w, unsigned short* __restrict__ wt,
                    int Cout, int Cin, int CoutPad) {
  int i = blockIdx.x * 256 + threadIdx.x;
  int K = 9 * Cin;
  if (i >= CoutPad * K) return;
  int co = i / K;
  int k  = i - co * K;
  int kidx = k / Cin;
  int ci   = k - kidx * Cin;
  unsigned short v = 0;
  if (co < Cout) v = f2bf(w[((long long)co * Cin + ci) * 9 + kidx]);
  wt[i] = v;
}

// ---------------------------------------------------------------------------
// Generic implicit-GEMM 3x3 conv, pad=1, NHWC bf16 in, bf16 or f32 NHWC out.
// Block: 256 thr = 8 waves (4 M x 2 N). Tile: M=128, N=64, Kstep=32.
// A tile staged with double-buffered GLOBAL_LOAD_ASYNC_TO_LDS_B128, one K-step
// ahead of the WMMA pipeline. cshift = log2(Cin/32).
// ---------------------------------------------------------------------------
__global__ __launch_bounds__(256)
void conv3x3_wmma(const unsigned short* __restrict__ in_nhwc, int Cin, int cshift,
                  const unsigned short* __restrict__ wt, int Ktot,
                  const float* __restrict__ bias, int Cout,
                  unsigned short* __restrict__ out_bf16,
                  float* __restrict__ out_f32) {
  __shared__ uint4 As[2 * 640];                  // 2 buffers, 80B row pitch

  const int t    = threadIdx.x;
  const int lane = t & 31;
  const int wv   = t >> 5;
  const int wm   = wv & 3;                       // M-wave 0..3
  const int wn   = wv >> 2;                      // N-wave 0..1
  const int m0   = blockIdx.x * 128;
  const int n0   = blockIdx.y * 64;
  const int h16  = lane >> 4;
  const int l16  = lane & 15;

  const int nchunk = Cin >> 5;
  const int nsteps = 9 << cshift;

  v8f acc[2][2];
  for (int i = 0; i < 2; ++i)
    for (int j = 0; j < 2; ++j)
      for (int e = 0; e < 8; ++e) acc[i][j][e] = 0.f;

  // staging role: one 32B half of one im2col row
  const int srow  = t >> 1;
  const int shalf = t & 1;
  const int sm    = m0 + srow;
  const int sb    = sm >> 12;
  const int sy    = (sm & 4095) >> 6;
  const int sx    = sm & 63;
  const unsigned ldsbase = (unsigned)(size_t)&As[0];

  bool needz[2] = {false, false};

  auto stage = [&](int s, int buf) {
    int kidx = s >> cshift;
    int k3   = kidx / 3;
    int ci0  = (s & (nchunk - 1)) << 5;
    int dy = k3 - 1, dx = (kidx - k3 * 3) - 1;
    int yy = sy + dy, xx = sx + dx;
    bool vld = ((unsigned)yy < 64u) && ((unsigned)xx < 64u);
    int yc = iclamp(yy, 0, 63);
    int xc = iclamp(xx, 0, 63);
    long long q = (((long long)sb) << 12) + (yc << 6) + xc;
    const unsigned short* g = in_nhwc + q * Cin + ci0 + shalf * 16;
    unsigned loff = ldsbase + (unsigned)((buf * 640 + srow * 5 + shalf * 2) * 16);
    // unconditional: every wave issues exactly 2 async instructions per stage
    async_b128(loff, g);
    async_b128_o16(loff, g);
    needz[buf] = !vld;
  };

  stage(0, 0);

  for (int s = 0; s < nsteps; ++s) {
    const int buf  = s & 1;
    const bool more = (s + 1) < nsteps;
    if (more) stage(s + 1, buf ^ 1);

    if (more) wait_async_le2();                  // previous buffer landed
    else      wait_async_0();

    if (needz[buf]) {                            // halo rows -> exact zeros
      uint4 z; z.x = z.y = z.z = z.w = 0u;
      As[buf * 640 + srow * 5 + shalf * 2 + 0] = z;
      As[buf * 640 + srow * 5 + shalf * 2 + 1] = z;
    }
    __syncthreads();

    const int bbase = buf * 640;
    const int k0 = s << 5;

    FragU fa[2], fb[2];
#pragma unroll
    for (int i = 0; i < 2; ++i) {
      int r = wm * 32 + i * 16 + l16;
      fa[i].q[0] = As[bbase + r * 5 + h16];      // K = h*8 .. h*8+7
      fa[i].q[1] = As[bbase + r * 5 + 2 + h16];  // K = 16+h*8 ..
    }
#pragma unroll
    for (int j = 0; j < 2; ++j) {
      int n = n0 + wn * 32 + j * 16 + l16;
      const uint4* bp = (const uint4*)(wt + (long long)n * Ktot + k0);
      fb[j].q[0] = bp[h16];                      // K = h*8 .. h*8+7
      fb[j].q[1] = bp[2 + h16];
    }
#pragma unroll
    for (int i = 0; i < 2; ++i)
#pragma unroll
      for (int j = 0; j < 2; ++j)
        acc[i][j] = __builtin_amdgcn_wmma_f32_16x16x32_bf16(
            false, fa[i].v, false, fb[j].v, (short)0, acc[i][j], false, false);

    __syncthreads();                             // reads done before buf reuse
  }

  // epilogue: bias + store NHWC
  for (int j = 0; j < 2; ++j) {
    int n = n0 + wn * 32 + j * 16 + l16;
    if (n >= Cout) continue;
    float bv = bias[n];
    for (int i = 0; i < 2; ++i) {
      for (int r = 0; r < 8; ++r) {
        int m = m0 + wm * 32 + i * 16 + h16 * 8 + r;
        float v = acc[i][j][r] + bv;
        if (out_bf16) out_bf16[(long long)m * Cout + n] = f2bf(v);
        else          out_f32[(long long)m * Cout + n]  = v;
      }
    }
  }
}

// ---------------------------------------------------------------------------
// Fused deformable conv: bilinear sample 64 pixels x 256 ch per kidx into
// LDS, then WMMA GEMM against w_def^T. Block: 256 thr = 8 waves (2M x 4N).
// Tile: M=64 pixels, N=256 out channels. Output NCHW f32.
// ---------------------------------------------------------------------------
__global__ __launch_bounds__(256)
void deform_wmma(const unsigned short* __restrict__ fused,  // [16384][512], vis @ +256
                 const float* __restrict__ offb,            // [16384][18]
                 const unsigned short* __restrict__ wdeft,  // [256][2304]
                 const float* __restrict__ bdef,
                 float* __restrict__ out) {                 // [4][256][4096]
  __shared__ uint4 Sm[64 * 33];                 // 528B row pitch (512 used)

  const int t    = threadIdx.x;
  const int lane = t & 31;
  const int wv   = t >> 5;
  const int wm   = wv & 1;                      // M-wave 0..1
  const int wn   = wv >> 1;                     // N-wave 0..3
  const int m0   = blockIdx.x * 64;
  const int h16  = lane >> 4;
  const int l16  = lane & 15;

  v8f acc[2][4];
  for (int i = 0; i < 2; ++i)
    for (int j = 0; j < 4; ++j)
      for (int e = 0; e < 8; ++e) acc[i][j][e] = 0.f;

  // sampling role: 4 threads per pixel, 64 channels each
  const int sp = t >> 2;
  const int sq = t & 3;
  const int sm = m0 + sp;
  const int sb = sm >> 12;
  const int sy = (sm & 4095) >> 6;
  const int sx = sm & 63;
  const long long pixbase = ((long long)sb) << 12;

  for (int kidx = 0; kidx < 9; ++kidx) {
    // --- bilinear sampling stage ---
    float dyv = offb[(long long)sm * 18 + 2 * kidx];
    float dxv = offb[(long long)sm * 18 + 2 * kidx + 1];
    float y = dyv + (float)(kidx / 3 - 1) + (float)sy;
    float x = dxv + (float)(kidx % 3 - 1) + (float)sx;
    float y0f = floorf(y), x0f = floorf(x);
    float ly = y - y0f, lx = x - x0f;
    int y0 = (int)y0f, x0 = (int)x0f;
    int y1 = y0 + 1,  x1 = x0 + 1;
    bool vy0 = (y0 >= 0 && y0 <= 63), vy1 = (y1 >= 0 && y1 <= 63);
    bool vx0 = (x0 >= 0 && x0 <= 63), vx1 = (x1 >= 0 && x1 <= 63);
    int cy0 = iclamp(y0, 0, 63), cy1 = iclamp(y1, 0, 63);
    int cx0 = iclamp(x0, 0, 63), cx1 = iclamp(x1, 0, 63);
    float w00 = (1.f - ly) * (1.f - lx) * ((vy0 && vx0) ? 1.f : 0.f);
    float w01 = (1.f - ly) * lx         * ((vy0 && vx1) ? 1.f : 0.f);
    float w10 = ly * (1.f - lx)         * ((vy1 && vx0) ? 1.f : 0.f);
    float w11 = ly * lx                 * ((vy1 && vx1) ? 1.f : 0.f);
    const unsigned short* p00 = fused + (pixbase + cy0 * 64 + cx0) * 512 + 256;
    const unsigned short* p01 = fused + (pixbase + cy0 * 64 + cx1) * 512 + 256;
    const unsigned short* p10 = fused + (pixbase + cy1 * 64 + cx0) * 512 + 256;
    const unsigned short* p11 = fused + (pixbase + cy1 * 64 + cx1) * 512 + 256;

    __syncthreads();                             // protect prior GEMM reads
#pragma unroll
    for (int it = 0; it < 8; ++it) {
      int ci = sq * 64 + it * 8;
      uint4 a00 = *(const uint4*)(p00 + ci);
      uint4 a01 = *(const uint4*)(p01 + ci);
      uint4 a10 = *(const uint4*)(p10 + ci);
      uint4 a11 = *(const uint4*)(p11 + ci);
      const unsigned* q00 = (const unsigned*)&a00;
      const unsigned* q01 = (const unsigned*)&a01;
      const unsigned* q10 = (const unsigned*)&a10;
      const unsigned* q11 = (const unsigned*)&a11;
      unsigned ow[4];
#pragma unroll
      for (int wd = 0; wd < 4; ++wd) {
        float flo = w00 * bf2f((unsigned short)(q00[wd] & 0xFFFFu))
                  + w01 * bf2f((unsigned short)(q01[wd] & 0xFFFFu))
                  + w10 * bf2f((unsigned short)(q10[wd] & 0xFFFFu))
                  + w11 * bf2f((unsigned short)(q11[wd] & 0xFFFFu));
        float fhi = w00 * bf2f((unsigned short)(q00[wd] >> 16))
                  + w01 * bf2f((unsigned short)(q01[wd] >> 16))
                  + w10 * bf2f((unsigned short)(q10[wd] >> 16))
                  + w11 * bf2f((unsigned short)(q11[wd] >> 16));
        ow[wd] = (unsigned)f2bf(flo) | ((unsigned)f2bf(fhi) << 16);
      }
      uint4 o; o.x = ow[0]; o.y = ow[1]; o.z = ow[2]; o.w = ow[3];
      Sm[sp * 33 + (ci >> 3)] = o;
    }
    __syncthreads();

    // --- GEMM over this kidx's 256-wide K block ---
    for (int ks = 0; ks < 8; ++ks) {
      FragU fa[2];
#pragma unroll
      for (int i = 0; i < 2; ++i) {
        int r = wm * 32 + i * 16 + l16;
        fa[i].q[0] = Sm[r * 33 + ks * 4 + h16];
        fa[i].q[1] = Sm[r * 33 + ks * 4 + 2 + h16];
      }
      const int kglob = kidx * 256 + ks * 32;
#pragma unroll
      for (int j = 0; j < 4; ++j) {
        int n = wn * 64 + j * 16 + l16;
        const uint4* bp = (const uint4*)(wdeft + (long long)n * 2304 + kglob);
        FragU fb;
        fb.q[0] = bp[h16];
        fb.q[1] = bp[2 + h16];
#pragma unroll
        for (int i = 0; i < 2; ++i)
          acc[i][j] = __builtin_amdgcn_wmma_f32_16x16x32_bf16(
              false, fa[i].v, false, fb.v, (short)0, acc[i][j], false, false);
      }
    }
  }

  // epilogue: bias + NCHW f32 store
  for (int j = 0; j < 4; ++j) {
    int n = wn * 64 + j * 16 + l16;
    float bv = bdef[n];
    for (int i = 0; i < 2; ++i) {
      for (int r = 0; r < 8; ++r) {
        int m = m0 + wm * 32 + i * 16 + h16 * 8 + r;
        int b = m >> 12, hw = m & 4095;
        out[(((long long)b * 256 + n) << 12) + hw] = acc[i][j][r] + bv;
      }
    }
  }
}

// ---------------------------------------------------------------------------
extern "C" void kernel_launch(void* const* d_in, const int* in_sizes, int n_in,
                              void* d_out, int out_size, void* d_ws, size_t ws_size,
                              hipStream_t stream) {
  (void)in_sizes; (void)n_in; (void)out_size; (void)ws_size;
  const float* feat_ir  = (const float*)d_in[0];
  const float* feat_vis = (const float*)d_in[1];
  const float* w1    = (const float*)d_in[2];
  const float* b1    = (const float*)d_in[3];
  const float* w2    = (const float*)d_in[4];
  const float* b2    = (const float*)d_in[5];
  const float* w_off = (const float*)d_in[6];
  const float* b_off = (const float*)d_in[7];
  const float* w_def = (const float*)d_in[8];
  const float* b_def = (const float*)d_in[9];
  float* out = (float*)d_out;

  size_t off = 0;
  auto take = [&](size_t bytes) -> void* {
    size_t o = (off + 255) & ~(size_t)255;
    off = o + bytes;
    return (void*)((char*)d_ws + o);
  };
  unsigned short* fused = (unsigned short*)take((size_t)16384 * 512 * 2);
  unsigned short* c1    = (unsigned short*)take((size_t)16384 * 128 * 2);
  unsigned short* c2    = (unsigned short*)take((size_t)16384 * 64 * 2);
  float*          offb  = (float*)take((size_t)16384 * 18 * 4);
  unsigned short* w1t   = (unsigned short*)take((size_t)128 * 4608 * 2);
  unsigned short* w2t   = (unsigned short*)take((size_t)64 * 1152 * 2);
  unsigned short* wofft = (unsigned short*)take((size_t)64 * 576 * 2);
  unsigned short* wdeft = (unsigned short*)take((size_t)256 * 2304 * 2);

  {
    int n = 16384 * 512;
    pack_fused_kernel<<<dim3((n + 255) / 256), 256, 0, stream>>>(feat_ir, feat_vis, fused);
  }
  { int n = 128 * 4608; pack_wt_kernel<<<dim3((n + 255) / 256), 256, 0, stream>>>(w1,    w1t,   128, 512, 128); }
  { int n = 64  * 1152; pack_wt_kernel<<<dim3((n + 255) / 256), 256, 0, stream>>>(w2,    w2t,    64, 128,  64); }
  { int n = 64  *  576; pack_wt_kernel<<<dim3((n + 255) / 256), 256, 0, stream>>>(w_off, wofft,  18,  64,  64); }
  { int n = 256 * 2304; pack_wt_kernel<<<dim3((n + 255) / 256), 256, 0, stream>>>(w_def, wdeft, 256, 256, 256); }

  // conv1: 512 -> 128 (bf16 out)
  conv3x3_wmma<<<dim3(128, 2), 256, 0, stream>>>(fused, 512, 4, w1t, 4608, b1, 128, c1, nullptr);
  // conv2: 128 -> 64 (bf16 out)
  conv3x3_wmma<<<dim3(128, 1), 256, 0, stream>>>(c1, 128, 2, w2t, 1152, b2, 64, c2, nullptr);
  // conv_off: 64 -> 18 (f32 out, padded weights)
  conv3x3_wmma<<<dim3(128, 1), 256, 0, stream>>>(c2, 64, 1, wofft, 576, b_off, 18, nullptr, offb);
  // fused deformable conv: 256 -> 256, NCHW f32 out
  deform_wmma<<<dim3(256), 256, 0, stream>>>(fused, offb, wdeft, b_def, out);
}
